// FullAttention_16509854286704
// MI455X (gfx1250) — matrix operand order repaired and checked by reading, not compiled, not run on hardware
//
#include <hip/hip_runtime.h>

#pragma clang diagnostic ignored "-Wdeprecated-builtins"

typedef _Float16 half_t;
typedef __attribute__((ext_vector_type(16))) _Float16 v16h;
typedef __attribute__((ext_vector_type(8)))  _Float16 v8h;
typedef __attribute__((ext_vector_type(2)))  _Float16 h2;
typedef __attribute__((ext_vector_type(8)))  float    v8f;
typedef __attribute__((ext_vector_type(4)))  float    f4;

// v_exp_f32 is exp2 on AMD; use the raw builtin (log2-domain softmax)
__device__ __forceinline__ float fast_exp2(float x) {
  return __builtin_amdgcn_exp2f(x);
}

// pack two f32 into one half2 with a single v_cvt_pk_rtz_f16_f32
__device__ __forceinline__ h2 pack_h2(float a, float b) {
  return __builtin_bit_cast(h2, __builtin_amdgcn_cvt_pkrtz(a, b));
}

// ds_swizzle group-of-32 encoding: offset[14:10]=xor, [9:5]=or, [4:0]=and
#define SWZI(x, imm) __builtin_amdgcn_ds_swizzle((x), (imm))

__device__ __forceinline__ h2 swz_h2(h2 x, const int imm_is) {
  // imm selected by constant index to keep the builtin's immediate a literal
  int i = __builtin_bit_cast(int, x);
  switch (imm_is) {
    case 0: i = SWZI(i, 0x041f); break;  // xor 1
    case 1: i = SWZI(i, 0x081f); break;  // xor 2
    case 2: i = SWZI(i, 0x101f); break;  // xor 4
    default: i = SWZI(i, 0x201f); break; // xor 8 (stays inside 16-lane half)
  }
  return __builtin_bit_cast(h2, i);
}

// lowers to v_pk_max_num_f16 (llvm.maxnum.v2f16)
__device__ __forceinline__ h2 max_h2(h2 a, h2 b) {
  return __builtin_elementwise_max(a, b);
}

// two 8-f16 chunks from LDS (p, p+16; 16B-aligned) -> one B fragment
__device__ __forceinline__ v16h lds_frag16(const half_t* p) {
  v8h lo = *(const v8h*)(p);
  v8h hi = *(const v8h*)(p + 16);
  return __builtin_shufflevector(lo, hi, 0,1,2,3,4,5,6,7,8,9,10,11,12,13,14,15);
}

constexpr int Bn = 2, Ln = 2048, Sn = 2048, Hn = 16, En = 64, Dn = 64;
constexpr int WAVES = 8;     // 256 threads / wave32
constexpr int QROWS = 16;    // query rows per wave
constexpr int KTILE = 64;    // keys per iteration (four 16-col tiles)
constexpr int PITCH = 72;    // LDS f16 row pitch: 36 dwords -> conflict-free 16-lane stride

__launch_bounds__(256)
__global__ void flash_attn_wmma(const float* __restrict__ Q,
                                const float* __restrict__ K,
                                const float* __restrict__ V,
                                float* __restrict__ O) {
  __shared__ half_t Kt[KTILE * PITCH];          // [s_local][e]     9216 B
  __shared__ half_t Vt[Dn * PITCH];             // [d][s_local]     9216 B (transposed)
  __shared__ half_t Pt[WAVES][QROWS * PITCH];   // per-wave P stage 18432 B

  const int tid     = threadIdx.x;
  const int wave    = tid >> 5;
  const int lane    = tid & 31;
  const int half_hi = (lane >> 4) & 1;
  const int ln      = lane & 15;
  const int kbo     = half_hi * 8;    // B-fragment K offset for hi lanes
  const int kh      = half_hi * 8;    // A-fragment K offset for hi lanes

  const int h  = blockIdx.y;
  const int b  = blockIdx.z;
  const int q0 = blockIdx.x * (WAVES * QROWS) + wave * QROWS;
  // fold log2(e) into the score scale so raw v_exp_f32 (exp2) computes softmax
  const float scale = 0.125f * 1.44269504088896340736f;

  const size_t rstride = (size_t)Hn * En;  // floats between consecutive s (== Hn*Dn)
  const float* kbasep = K + ((size_t)b * Sn * Hn + h) * (size_t)En;
  const float* vbasep = V + ((size_t)b * Sn * Hn + h) * (size_t)Dn;

  // ---- Q as two 16x32 f16 A-fragments, pre-scaled by scale*log2(e)
  v16h qa[2];
  {
    const float* qrow = Q + (((size_t)b * Ln + (q0 + ln)) * Hn + h) * (size_t)En;
#pragma unroll
    for (int f = 0; f < 2; ++f) {
      f4 c0 = *(const f4*)(qrow + 32 * f + kh);
      f4 c1 = *(const f4*)(qrow + 32 * f + kh + 4);
      f4 c2 = *(const f4*)(qrow + 32 * f + 16 + kh);
      f4 c3 = *(const f4*)(qrow + 32 * f + 16 + kh + 4);
#pragma unroll
      for (int j = 0; j < 4; ++j) {
        qa[f][j]      = (half_t)(c0[j] * scale);
        qa[f][4 + j]  = (half_t)(c1[j] * scale);
        qa[f][8 + j]  = (half_t)(c2[j] * scale);
        qa[f][12 + j] = (half_t)(c3[j] * scale);
      }
    }
  }

  v16h ones;
#pragma unroll
  for (int j = 0; j < 16; ++j) ones[j] = (half_t)1.0f;

  v8f   acc[4] = {};   // 16x64 output accumulator
  v8f   accL   = {};   // softmax denominator via P @ ones (replicated across lanes)
  float m_r[8];        // running max, log2 domain
#pragma unroll
  for (int r = 0; r < 8; ++r) m_r[r] = -1e30f;

  // staging roles
  const int srow = tid >> 2;          // K: 4 threads per key row
  const int scol = (tid & 3) * 16;
  const int vk0  = (tid >> 3) * 2;    // V: 2 consecutive k rows ...
  const int vd0  = (tid & 7) * 8;     // ... x 8 d columns per thread

  for (int s0 = 0; s0 < Sn; s0 += KTILE) {
    __syncthreads();  // previous iteration's fragment reads done before overwrite

    // ---- stage K tile (f32 -> f16), row-major [s][e]
    {
      const float* kp = kbasep + (size_t)(s0 + srow) * rstride + scol;
      f4 a0 = *(const f4*)(kp);
      f4 a1 = *(const f4*)(kp + 4);
      f4 a2 = *(const f4*)(kp + 8);
      f4 a3 = *(const f4*)(kp + 12);
      v8h h0, h1;
#pragma unroll
      for (int j = 0; j < 4; ++j) {
        h0[j]     = (half_t)a0[j];  h0[4 + j] = (half_t)a1[j];
        h1[j]     = (half_t)a2[j];  h1[4 + j] = (half_t)a3[j];
      }
      half_t* dst = &Kt[srow * PITCH + scol];
      *(v8h*)dst       = h0;
      *(v8h*)(dst + 8) = h1;
      if (s0 + KTILE < Sn) __builtin_prefetch(kp + KTILE * rstride, 0, 0);
    }
    // ---- stage V tile transposed [d][s_local]; pack 2 k-values per half2 store
    {
      const float* vp0 = vbasep + (size_t)(s0 + vk0) * rstride + vd0;
      const float* vp1 = vp0 + rstride;
      f4 a0 = *(const f4*)(vp0);
      f4 a1 = *(const f4*)(vp0 + 4);
      f4 b0 = *(const f4*)(vp1);
      f4 b1 = *(const f4*)(vp1 + 4);
#pragma unroll
      for (int j = 0; j < 4; ++j) {
        *(h2*)&Vt[(vd0 + j) * PITCH + vk0]     = pack_h2(a0[j], b0[j]);
        *(h2*)&Vt[(vd0 + 4 + j) * PITCH + vk0] = pack_h2(a1[j], b1[j]);
      }
      if (s0 + KTILE < Sn) __builtin_prefetch(vp0 + KTILE * rstride, 0, 0);
    }
    __syncthreads();

    // ---- scores = (Q*scale*log2e) @ K^T : four 16x16 f32 tiles
    v8f sc[4];
#pragma unroll
    for (int ct = 0; ct < 4; ++ct) {
      const half_t* krow = &Kt[(ct * 16 + ln) * PITCH + kbo];
      v16h b0 = lds_frag16(krow);        // e 0..31 chunk
      v16h b1 = lds_frag16(krow + 32);   // e 32..63 chunk
      v8f c = {};
      c = __builtin_amdgcn_wmma_f32_16x16x32_f16(false, qa[0], false, b0,
                                                 (short)0, c, false, false);
      c = __builtin_amdgcn_wmma_f32_16x16x32_f16(false, qa[1], false, b1,
                                                 (short)0, c, false, false);
      sc[ct] = c;
    }

    // ---- row max: in-lane max over 4 tiles, then packed-f16 butterfly over 16 lanes
    h2 hp[4];
#pragma unroll
    for (int i = 0; i < 4; ++i) {
      const int r0 = 2 * i, r1 = 2 * i + 1;
      float a  = fmaxf(fmaxf(sc[0][r0], sc[1][r0]), fmaxf(sc[2][r0], sc[3][r0]));
      float b2 = fmaxf(fmaxf(sc[0][r1], sc[1][r1]), fmaxf(sc[2][r1], sc[3][r1]));
      hp[i] = pack_h2(a, b2);            // one v_cvt_pk_rtz_f16_f32
    }
#pragma unroll
    for (int st = 0; st < 4; ++st)
#pragma unroll
      for (int i = 0; i < 4; ++i)
        hp[i] = max_h2(hp[i], swz_h2(hp[i], st));  // v_pk_max_num_f16

    // ---- online softmax (row = r + 8*half); row-sum comes from WMMA below
#pragma unroll
    for (int r = 0; r < 8; ++r) {
      float rmax = (float)hp[r >> 1][r & 1];
      float mnew = fmaxf(m_r[r], rmax);
      float corr = fast_exp2(m_r[r] - mnew);   // v_exp_f32 directly
      m_r[r] = mnew;
#pragma unroll
      for (int ct = 0; ct < 4; ++ct) sc[ct][r] = fast_exp2(sc[ct][r] - mnew);
#pragma unroll
      for (int t = 0; t < 4; ++t) acc[t][r] *= corr;
      accL[r] *= corr;
    }

    // ---- P: C-layout -> A-layout transpose via per-wave LDS (hw in-order same-wave)
    half_t* prow = Pt[wave];
#pragma unroll
    for (int ct = 0; ct < 4; ++ct)
#pragma unroll
      for (int r = 0; r < 8; ++r)
        prow[(r + 8 * half_hi) * PITCH + ct * 16 + ln] = (half_t)sc[ct][r];
    __builtin_amdgcn_wave_barrier();

    v16h pa[2];
#pragma unroll
    for (int f = 0; f < 2; ++f) {
      const half_t* p1 = prow + ln * PITCH + 32 * f + kh;
      v8h lo = *(const v8h*)(p1);        // k chunk1
      v8h hi = *(const v8h*)(p1 + 16);   // k chunk2 (+16 halfs)
      pa[f] = __builtin_shufflevector(lo, hi, 0,1,2,3,4,5,6,7,8,9,10,11,12,13,14,15);
    }

    // ---- acc += P @ V ; accL += P @ ones (row sums, lanes replicated)
#pragma unroll
    for (int t = 0; t < 4; ++t) {
      const half_t* vrow = &Vt[(t * 16 + ln) * PITCH + kbo];
      v16h vb0 = lds_frag16(vrow);        // k 0..31 chunk
      v16h vb1 = lds_frag16(vrow + 32);   // k 32..63 chunk
      acc[t] = __builtin_amdgcn_wmma_f32_16x16x32_f16(false, pa[0], false, vb0,
                                                      (short)0, acc[t], false, false);
      acc[t] = __builtin_amdgcn_wmma_f32_16x16x32_f16(false, pa[1], false, vb1,
                                                      (short)0, acc[t], false, false);
    }
    accL = __builtin_amdgcn_wmma_f32_16x16x32_f16(false, pa[0], false, ones,
                                                  (short)0, accL, false, false);
    accL = __builtin_amdgcn_wmma_f32_16x16x32_f16(false, pa[1], false, ones,
                                                  (short)0, accL, false, false);
  }

  // ---- epilogue: O[b, q0+m, h, d] = acc / l
#pragma unroll
  for (int r = 0; r < 8; ++r) {
    const float inv = 1.0f / accL[r];
    float* orow = O + (((size_t)b * Ln + (q0 + r + 8 * half_hi)) * Hn + h) * (size_t)Dn;
#pragma unroll
    for (int t = 0; t < 4; ++t) {
      orow[t * 16 + ln] = acc[t][r] * inv;
    }
  }
}

extern "C" void kernel_launch(void* const* d_in, const int* in_sizes, int n_in,
                              void* d_out, int out_size, void* d_ws, size_t ws_size,
                              hipStream_t stream) {
  const float* Q = (const float*)d_in[0];
  const float* K = (const float*)d_in[1];
  const float* V = (const float*)d_in[2];
  float*       O = (float*)d_out;
  (void)in_sizes; (void)n_in; (void)out_size; (void)d_ws; (void)ws_size;

  dim3 grid(Ln / (WAVES * QROWS), Hn, Bn);   // (16, 16, 2)
  dim3 block(WAVES * 32);                    // 256 threads = 8 wave32
  hipLaunchKernelGGL(flash_attn_wmma, grid, block, 0, stream, Q, K, V, O);
}
